// LSTMClassifier_51505247814207
// MI455X (gfx1250) — matrix operand order repaired and checked by reading, not compiled, not run on hardware
//
#include <hip/hip_runtime.h>
#include <stdint.h>

typedef __attribute__((ext_vector_type(16))) __bf16 v16bf;
typedef __attribute__((ext_vector_type(8)))  float  v8f;

namespace {
constexpr int kT  = 128;   // timesteps
constexpr int kF  = 64;    // input features (layer0)
constexpr int kH  = 512;   // hidden
constexpr int kG  = 2048;  // 4*H gate width
constexpr int kBT = 16;    // batch tile = WMMA M
constexpr int kGW = 2052;  // gates LDS row stride (floats), padded
constexpr int kHW = 520;   // h LDS row stride (bf16), padded
constexpr int kCW = 520;   // c LDS row stride (floats), padded
constexpr int kThreads = 512;   // 16 waves -> 4 waves/SIMD, ~256 VGPR cap/wave
}

union AB32 {                 // one 16x32 bf16 A/B fragment (16 bf16 per lane)
    v16bf v;
    unsigned short u[16];
    uint4 q[2];
};
union C8 {                   // one 16x16 f32 C/D fragment (8 f32 per lane)
    v8f   v;
    float f[8];
};

// float -> bf16 bits via the native gfx1250 convert (round-to-nearest-even).
__device__ __forceinline__ unsigned short f2bf(float f) {
    union { __bf16 b; unsigned short u; } cv;
    cv.b = (__bf16)f;
    return cv.u;
}

// gfx1250 has a hardware V_TANH_F32 (TRANS op); use it if the builtin exists.
__device__ __forceinline__ float fast_tanh(float x) {
#if __has_builtin(__builtin_amdgcn_tanhf)
    return __builtin_amdgcn_tanhf(x);
#else
    return tanhf(x);
#endif
}
// sigmoid(x) = 0.5*tanh(x/2) + 0.5 : one TRANS op instead of exp+rcp.
__device__ __forceinline__ float fast_sigmoid(float x) {
    return __builtin_fmaf(0.5f, fast_tanh(0.5f * x), 0.5f);
}

__device__ __forceinline__ v8f wmma_bf16(v16bf a, v16bf b, v8f c) {
    return __builtin_amdgcn_wmma_f32_16x16x32_bf16(
        /*neg_a=*/false, a, /*neg_b=*/false, b,
        /*c_mod=*/(short)0, c, /*reuse_a=*/false, /*reuse_b=*/false);
}

// A fragment (16x32 bf16, M=16 rows) from an LDS h-buffer, row stride kHW.
__device__ __forceinline__ v16bf load_a_lds(const unsigned short* h, int k0, int lane) {
    const int m = lane & 15, hi = lane >> 4;
    const unsigned short* p = h + m * kHW + k0 + 8 * hi;
    AB32 a;
    a.q[0] = *(const uint4*)(p);
    a.q[1] = *(const uint4*)(p + 16);
    return a.v;
}

// A fragment from global fp32 X[b][t][0..63] (layer0 input), converted to bf16.
__device__ __forceinline__ v16bf load_a_x(const float* __restrict__ X,
                                          int b0, int t, int kx, int lane) {
    const int m = lane & 15, hi = lane >> 4;
    const float* p = X + (((size_t)(b0 + m) * kT + t) * kF) + kx + 8 * hi;
    AB32 a;
#pragma unroll
    for (int i = 0; i < 8; ++i) {
        a.u[i]     = f2bf(p[i]);
        a.u[8 + i] = f2bf(p[16 + i]);
    }
    return a.v;
}

// B fragment (32x16 bf16) from transposed bf16 weights wt[K][2048].
__device__ __forceinline__ AB32 load_b_frag(const unsigned short* __restrict__ wt,
                                            int k, int n0) {
    const int l = k & 31, hi = l >> 4;
    const unsigned short* p = wt + (size_t)(k - l + 8 * hi) * kG + n0 + (l & 15);
    AB32 b;
#pragma unroll
    for (int i = 0; i < 8; ++i) {
        b.u[i]     = p[(size_t)i * kG];
        b.u[8 + i] = p[(size_t)(16 + i) * kG];
    }
    return b;
}

// K=512 GEMM segment: acc[8] += A(hbuf) * B(wt).
// ks loop stays rolled (unroll 1) so the scheduler cannot hoist loads across
// iterations and spill. Within a step, all 128 16-bit global loads are batched
// into b[8] before the 8 WMMAs; loads return in order, so the compiler emits
// partial s_wait_loadcnt and overlaps loads with WMMAs.
__device__ __forceinline__ void gemm_seg(C8* acc, const unsigned short* hbuf,
                                         const unsigned short* __restrict__ wt,
                                         int nbase, int lane) {
#pragma unroll 1
    for (int ks = 0; ks < 16; ++ks) {
        const int k0 = ks << 5;
        AB32 b[8];
#pragma unroll
        for (int nt = 0; nt < 8; ++nt)
            b[nt] = load_b_frag(wt, k0 + lane, nbase + (nt << 4));
        v16bf a = load_a_lds(hbuf, k0, lane);
#pragma unroll
        for (int nt = 0; nt < 8; ++nt) acc[nt].v = wmma_bf16(a, b[nt].v, acc[nt].v);
    }
}

// Scatter 8 accumulator tiles into the LDS gate matrix (fp32).
__device__ __forceinline__ void store_gates(float* gates, const C8* acc,
                                            int nbase, int lane) {
    const int col = nbase + (lane & 15);
    const int hi  = lane >> 4;
#pragma unroll
    for (int nt = 0; nt < 8; ++nt) {
#pragma unroll
        for (int r = 0; r < 8; ++r)
            gates[(r + 8 * hi) * kGW + col + (nt << 4)] = acc[nt].f[r];
    }
}

// Per-thread LSTM cell update for 16 (m, j) pairs, j interleaved by 32 so that
// consecutive lanes touch consecutive LDS words (bank-conflict free).
// Cell state c lives in LDS. All activations use the HW tanh unit.
__device__ __forceinline__ void cell_update(float* gates, unsigned short* hbuf,
                                            float* cls, int m, int jl,
                                            bool dump_h_fp32) {
#pragma unroll
    for (int jj = 0; jj < 16; ++jj) {
        const int j = jl + (jj << 5);
        float ig = fast_sigmoid(gates[m * kGW + j]);
        float fg = fast_sigmoid(gates[m * kGW + j + 512]);
        float gg = fast_tanh  (gates[m * kGW + j + 1024]);
        float og = fast_sigmoid(gates[m * kGW + j + 1536]);
        float c = fg * cls[m * kCW + j] + ig * gg;
        cls[m * kCW + j] = c;
        float hv = og * fast_tanh(c);
        hbuf[m * kHW + j] = f2bf(hv);
        if (dump_h_fp32) gates[m * kGW + j] = hv;  // column j owned by this thread
    }
}

// ---------------- prep kernels ----------------

// src is [R][C] fp32 row-major; dst is [C][R] bf16 (R = 2048 gate dim).
__global__ void transpose_to_bf16(const float* __restrict__ src,
                                  unsigned short* __restrict__ dst, int R, int C) {
    int idx = blockIdx.x * blockDim.x + threadIdx.x;
    if (idx >= R * C) return;
    int r = idx / C, c = idx % C;
    dst[(size_t)c * R + r] = f2bf(src[idx]);
}

__global__ void bias_sum(const float* __restrict__ a, const float* __restrict__ b,
                         float* __restrict__ o, int n) {
    int i = blockIdx.x * blockDim.x + threadIdx.x;
    if (i < n) o[i] = a[i] + b[i];
}

// ---------------- fused 2-layer LSTM + classifier ----------------
// grid = B/16 workgroups; block = 512 threads (16 waves); one batch tile each.
__global__ void __launch_bounds__(kThreads)
lstm2_fused(const float* __restrict__ X,
            const unsigned short* __restrict__ wih0t,   // [64][2048]  bf16
            const unsigned short* __restrict__ whh0t,   // [512][2048] bf16
            const unsigned short* __restrict__ wih1t,   // [512][2048] bf16
            const unsigned short* __restrict__ whh1t,   // [512][2048] bf16
            const float* __restrict__ b0sum,            // [2048]
            const float* __restrict__ b1sum,            // [2048]
            const float* __restrict__ Wc,               // [3][512]
            const float* __restrict__ bc,               // [3]
            float* __restrict__ out)                    // [B][3]
{
    extern __shared__ char smem[];
    float*          gates = (float*)smem;                        // 16 x kGW fp32
    unsigned short* h0ls  = (unsigned short*)(gates + 16 * kGW); // 16 x kHW bf16
    unsigned short* h1ls  = h0ls + 16 * kHW;                     // 16 x kHW bf16
    float*          c0ls  = (float*)(h1ls + 16 * kHW);           // 16 x kCW fp32
    float*          c1ls  = c0ls + 16 * kCW;                     // 16 x kCW fp32

    const int tid  = threadIdx.x;
    const int lane = tid & 31;
    const int wv   = tid >> 5;          // wave 0..15
    const int b0   = blockIdx.x * kBT;  // batch tile base
    const int nbase = wv << 7;          // this wave's 128 gate columns

    // zero hidden + cell state buffers
    for (int i = tid; i < 16 * kHW; i += kThreads) { h0ls[i] = 0; h1ls[i] = 0; }
    for (int i = tid; i < 16 * kCW; i += kThreads) { c0ls[i] = 0.0f; c1ls[i] = 0.0f; }

    // per-thread elementwise ownership: m row, lane-interleaved j
    const int em = tid >> 5;        // 0..15
    const int jl = tid & 31;        // j = jl + 32*jj, jj = 0..15

    // biases are loop-invariant: load once per wave/lane
    float bias0v[8], bias1v[8];
#pragma unroll
    for (int nt = 0; nt < 8; ++nt) {
        bias0v[nt] = b0sum[nbase + (nt << 4) + (lane & 15)];
        bias1v[nt] = b1sum[nbase + (nt << 4) + (lane & 15)];
    }

    __syncthreads();

#pragma unroll 1
    for (int t = 0; t < kT; ++t) {
        // ===== layer 0: gates = b0 + h0_prev @ Whh0^T + x_t @ Wih0^T =====
        C8 acc[8];
#pragma unroll
        for (int nt = 0; nt < 8; ++nt)
#pragma unroll
            for (int r = 0; r < 8; ++r) acc[nt].f[r] = bias0v[nt];

        gemm_seg(acc, h0ls, whh0t, nbase, lane);      // K = 512 (recurrent)

#pragma unroll 1                                       // K = 64 (input), 2 k-steps
        for (int kxs = 0; kxs < 2; ++kxs) {
            const int kx = kxs << 5;
            AB32 b[8];
#pragma unroll
            for (int nt = 0; nt < 8; ++nt)
                b[nt] = load_b_frag(wih0t, kx + lane, nbase + (nt << 4));
            v16bf a = load_a_x(X, b0, t, kx, lane);
#pragma unroll
            for (int nt = 0; nt < 8; ++nt) acc[nt].v = wmma_bf16(a, b[nt].v, acc[nt].v);
        }

        store_gates(gates, acc, nbase, lane);
        __syncthreads();
        cell_update(gates, h0ls, c0ls, em, jl, false);
        __syncthreads();

        // ===== layer 1: gates = b1 + h1_prev @ Whh1^T + h0_cur @ Wih1^T =====
#pragma unroll
        for (int nt = 0; nt < 8; ++nt)
#pragma unroll
            for (int r = 0; r < 8; ++r) acc[nt].f[r] = bias1v[nt];

        gemm_seg(acc, h1ls, whh1t, nbase, lane);      // recurrent K=512
        gemm_seg(acc, h0ls, wih1t, nbase, lane);      // input     K=512
        store_gates(gates, acc, nbase, lane);
        __syncthreads();
        cell_update(gates, h1ls, c1ls, em, jl, t == kT - 1);  // last step: fp32 h -> gates
        __syncthreads();
    }

    // ===== classifier on final h1 (fp32 copy lives in gates[:, 0:512]) =====
    if (tid < kBT * 3) {
        const int m = tid / 3, cls = tid % 3;
        float s = 0.0f;
        for (int k = 0; k < kH; ++k)
            s += gates[m * kGW + k] * Wc[cls * kH + k];
        out[(size_t)(b0 + m) * 3 + cls] = s + bc[cls];
    }
}

// ---------------- host launcher ----------------

extern "C" void kernel_launch(void* const* d_in, const int* in_sizes, int n_in,
                              void* d_out, int out_size, void* d_ws, size_t ws_size,
                              hipStream_t stream) {
    (void)in_sizes; (void)n_in; (void)out_size; (void)ws_size;

    const float* X     = (const float*)d_in[0];
    const float* W_ih0 = (const float*)d_in[1];
    const float* W_hh0 = (const float*)d_in[2];
    const float* b_ih0 = (const float*)d_in[3];
    const float* b_hh0 = (const float*)d_in[4];
    const float* W_ih1 = (const float*)d_in[5];
    const float* W_hh1 = (const float*)d_in[6];
    const float* b_ih1 = (const float*)d_in[7];
    const float* b_hh1 = (const float*)d_in[8];
    const float* Wc    = (const float*)d_in[9];
    const float* bc    = (const float*)d_in[10];
    float* out = (float*)d_out;

    // workspace layout (bf16 transposed weights + fused biases, ~6.6 MB)
    unsigned short* ws    = (unsigned short*)d_ws;
    unsigned short* wih0t = ws;                          // 64*2048
    unsigned short* whh0t = wih0t + (size_t)kF * kG;     // 512*2048
    unsigned short* wih1t = whh0t + (size_t)kH * kG;     // 512*2048
    unsigned short* whh1t = wih1t + (size_t)kH * kG;     // 512*2048
    float* b0s = (float*)(whh1t + (size_t)kH * kG);
    float* b1s = b0s + kG;

    const int thr = 256;
    transpose_to_bf16<<<(kG * kF + thr - 1) / thr, thr, 0, stream>>>(W_ih0, wih0t, kG, kF);
    transpose_to_bf16<<<(kG * kH + thr - 1) / thr, thr, 0, stream>>>(W_hh0, whh0t, kG, kH);
    transpose_to_bf16<<<(kG * kH + thr - 1) / thr, thr, 0, stream>>>(W_ih1, wih1t, kG, kH);
    transpose_to_bf16<<<(kG * kH + thr - 1) / thr, thr, 0, stream>>>(W_hh1, whh1t, kG, kH);
    bias_sum<<<(kG + thr - 1) / thr, thr, 0, stream>>>(b_ih0, b_hh0, b0s, kG);
    bias_sum<<<(kG + thr - 1) / thr, thr, 0, stream>>>(b_ih1, b_hh1, b1s, kG);

    const size_t smemBytes = (size_t)16 * kGW * sizeof(float)
                           + (size_t)2 * 16 * kHW * sizeof(unsigned short)
                           + (size_t)2 * 16 * kCW * sizeof(float);
    lstm2_fused<<<dim3(512 / kBT), dim3(kThreads), smemBytes, stream>>>(
        X, wih0t, whh0t, wih1t, whh1t, b0s, b1s, Wc, bc, out);
}
